// ResidualVectorQuantizer_27573690040474
// MI455X (gfx1250) — matrix operand order repaired and checked by reading, not compiled, not run on hardware
//
#include <hip/hip_runtime.h>
#include <hip/hip_bf16.h>

// Residual VQ on MI455X (gfx1250): split-bf16 WMMA score GEMM + in-register argmax.
// x: [8, 4096, 256] f32, codebooks: [7, 2048, 256] f32
// out: indices [7, 8, 4096] int32 ++ quantized [8, 4096, 256] f32 (flat in d_out)

#define B_   8
#define N_   4096
#define D_   256
#define K_   2048
#define Q_   7
#define BN   (B_ * N_)        // 32768 rows
#define LDS_STR 264           // bf16 elements per padded 256-wide row

typedef __attribute__((ext_vector_type(16))) __bf16 v16bf;
typedef __attribute__((ext_vector_type(8)))  float  v8f;
typedef __attribute__((ext_vector_type(8)))  __bf16 bf16x8;

// ---------------- helper kernels ----------------

__global__ void rvq_copy4(const float4* __restrict__ src, float4* __restrict__ dst, int n4) {
  int i = blockIdx.x * blockDim.x + threadIdx.x;
  if (i < n4) dst[i] = src[i];
}

// cnorm[qk] = 0.5 * |c_qk|^2 ; one wave32 per codebook row
__global__ void rvq_cnorm(const float* __restrict__ cb, float* __restrict__ cnorm) {
  int gw   = (blockIdx.x * blockDim.x + threadIdx.x) >> 5;
  int lane = threadIdx.x & 31;
  if (gw >= Q_ * K_) return;
  const float* row = cb + (size_t)gw * D_;
  float4 a = *(const float4*)(row + lane * 8);
  float4 b = *(const float4*)(row + lane * 8 + 4);
  float s = a.x*a.x + a.y*a.y + a.z*a.z + a.w*a.w
          + b.x*b.x + b.y*b.y + b.z*b.z + b.w*b.w;
#pragma unroll
  for (int t = 16; t >= 1; t >>= 1) s += __shfl_xor(s, t, 32);
  if (lane == 0) cnorm[gw] = 0.5f * s;
}

// quant = x - residual  (residual was accumulated in the quant buffer; rewrite in place)
__global__ void rvq_finalize(const float4* __restrict__ x, float4* __restrict__ q, int n4) {
  int i = blockIdx.x * blockDim.x + threadIdx.x;
  if (i >= n4) return;
  float4 xv = x[i], rv = q[i];
  rv.x = xv.x - rv.x; rv.y = xv.y - rv.y; rv.z = xv.z - rv.z; rv.w = xv.w - rv.w;
  q[i] = rv;
}

// ---------------- per-stage fused kernel ----------------
// Block: 256 threads (8 wave32), 128 rows. Each wave owns 16 rows, sweeps all K codes.
__global__ __launch_bounds__(256)
void rvq_stage(float* __restrict__ res,            // [BN, D] residual (read + update)
               const float* __restrict__ cb,       // [K, D] this stage's codebook
               const float* __restrict__ cnorm,    // [K] 0.5*|c|^2
               int* __restrict__ out_idx)          // [BN] indices for this stage
{
  __shared__ __bf16 lds_hi[32 * LDS_STR];          // staged codebook, bf16 hi
  __shared__ __bf16 lds_lo[32 * LDS_STR];          // staged codebook, bf16 lo
  __shared__ __bf16 lds_alo[8 * 16 * LDS_STR];     // per-wave residual lo-correction rows

  const int tid   = threadIdx.x;
  const int lane  = tid & 31;
  const int wave  = tid >> 5;
  const int half  = lane >> 4;   // which 16-lane half
  const int l16   = lane & 15;
  const int rbase = blockIdx.x * 128 + wave * 16;

  // ---- A fragments: residual rows rbase..rbase+15, split bf16.
  // hi part stays in VGPRs; lo part goes to a wave-private LDS region (canonical
  // row layout) and is re-loaded per d-chunk, freeing 64 VGPRs for load pipelining.
  // ISA 16-bit A 16x32 layout: lanes 0-15 hold K{0..7,16..23}; lanes 16-31 hold K{8..15,24..31}.
  v16bf a_hi[8];
  {
    const float* ar = res + (size_t)(rbase + l16) * D_;
    __bf16* alr = &lds_alo[(wave * 16 + l16) * LDS_STR];
#pragma unroll
    for (int dc = 0; dc < 8; ++dc) {
      const float* p = ar + dc * 32 + half * 8;
      float4 x0 = *(const float4*)(p + 0);
      float4 x1 = *(const float4*)(p + 4);
      float4 y0 = *(const float4*)(p + 16);
      float4 y1 = *(const float4*)(p + 20);
      float v[16] = {x0.x,x0.y,x0.z,x0.w, x1.x,x1.y,x1.z,x1.w,
                     y0.x,y0.y,y0.z,y0.w, y1.x,y1.y,y1.z,y1.w};
      bf16x8 lo0, lo1;
#pragma unroll
      for (int e = 0; e < 16; ++e) {
        __bf16 h = (__bf16)v[e];
        a_hi[dc][e] = h;
        __bf16 lb = (__bf16)(v[e] - (float)h);
        if (e < 8) lo0[e] = lb; else lo1[e - 8] = lb;
      }
      // canonical [row][d] layout: this lane covers d = dc*32 + half*8 + {0..7} and +16
      *(bf16x8*)(alr + dc * 32 + half * 8)      = lo0;
      *(bf16x8*)(alr + dc * 32 + half * 8 + 16) = lo1;
    }
  }
  // lds_alo is wave-private and same-wave LDS ops are in-order: no barrier needed.

  float best[8];
  int   bidx[8];
#pragma unroll
  for (int j = 0; j < 8; ++j) { best[j] = -3.4e38f; bidx[j] = 0; }

  for (int kb = 0; kb < K_; kb += 32) {
    __syncthreads();
    // ---- stage 32 codebook rows -> LDS as split bf16 (converted once, used by 8 waves)
    {
      const int row = tid >> 3;         // 0..31
      const int d0  = (tid & 7) * 32;   // 0..224
      const float* p = cb + (size_t)(kb + row) * D_ + d0;
      // prefetch next k-tile's chunk toward the WGP while we convert this one
      if (kb + 32 < K_) __builtin_prefetch(p + 32 * D_, 0, 1);
      bf16x8 h[4], l[4];
#pragma unroll
      for (int s = 0; s < 4; ++s) {
        float4 u0 = *(const float4*)(p + s * 8 + 0);
        float4 u1 = *(const float4*)(p + s * 8 + 4);
        float v[8] = {u0.x,u0.y,u0.z,u0.w, u1.x,u1.y,u1.z,u1.w};
#pragma unroll
        for (int e = 0; e < 8; ++e) {
          __bf16 hb = (__bf16)v[e];
          h[s][e] = hb;
          l[s][e] = (__bf16)(v[e] - (float)hb);
        }
      }
#pragma unroll
      for (int s = 0; s < 4; ++s) {
        *(bf16x8*)&lds_hi[row * LDS_STR + d0 + s * 8] = h[s];
        *(bf16x8*)&lds_lo[row * LDS_STR + d0 + s * 8] = l[s];
      }
    }
    __syncthreads();

    // ---- both 16-wide code subtiles; 4 independent accumulator chains and
    // explicitly double-buffered fragment loads (issue dc+1 loads before dc's WMMAs).
    // a_lo is issued FIRST in each batch: DScnt completes in-order, so the waits
    // for the B fragments (issued after it) subsume the a_lo wait -- no dscnt-0 drain.
    v8f c0h = {}, c0l = {}, c1h = {}, c1l = {};
    v16bf Bf[2][5];   // [buf]: 0=b0h 1=b0l 2=b1h 3=b1l 4=a_lo fragment

#define LOAD_FRAGS(dcv, buf)                                                    \
    do {                                                                        \
      const __bf16* ap_ = &lds_alo[(wave * 16 + l16) * LDS_STR                  \
                                   + (dcv) * 32 + half * 8];                    \
      union { v16bf v; bf16x8 h[2]; } u_;                                       \
      u_.h[0] = *(const bf16x8*)ap_;                                            \
      u_.h[1] = *(const bf16x8*)(ap_ + 16);                                     \
      Bf[buf][4] = u_.v;                                                        \
      const int boff_ = l16 * LDS_STR + (dcv) * 32 + half * 8;                  \
      u_.h[0] = *(const bf16x8*)&lds_hi[boff_];                                 \
      u_.h[1] = *(const bf16x8*)&lds_hi[boff_ + 16];                            \
      Bf[buf][0] = u_.v;                                                        \
      u_.h[0] = *(const bf16x8*)&lds_lo[boff_];                                 \
      u_.h[1] = *(const bf16x8*)&lds_lo[boff_ + 16];                            \
      Bf[buf][1] = u_.v;                                                        \
      u_.h[0] = *(const bf16x8*)&lds_hi[16 * LDS_STR + boff_];                  \
      u_.h[1] = *(const bf16x8*)&lds_hi[16 * LDS_STR + boff_ + 16];             \
      Bf[buf][2] = u_.v;                                                        \
      u_.h[0] = *(const bf16x8*)&lds_lo[16 * LDS_STR + boff_];                  \
      u_.h[1] = *(const bf16x8*)&lds_lo[16 * LDS_STR + boff_ + 16];             \
      Bf[buf][3] = u_.v;                                                        \
    } while (0)

    LOAD_FRAGS(0, 0);
#pragma unroll
    for (int dc = 0; dc < 8; ++dc) {
      const int cur = dc & 1;
      if (dc < 7) LOAD_FRAGS(dc + 1, cur ^ 1);
      c0h = __builtin_amdgcn_wmma_f32_16x16x32_bf16(false, a_hi[dc], false, Bf[cur][0], (short)0, c0h, false, false);
      c1h = __builtin_amdgcn_wmma_f32_16x16x32_bf16(false, a_hi[dc], false, Bf[cur][2], (short)0, c1h, false, false);
      c0l = __builtin_amdgcn_wmma_f32_16x16x32_bf16(false, a_hi[dc], false, Bf[cur][1], (short)0, c0l, false, false);
      c1l = __builtin_amdgcn_wmma_f32_16x16x32_bf16(false, a_hi[dc], false, Bf[cur][3], (short)0, c1l, false, false);
      c0l = __builtin_amdgcn_wmma_f32_16x16x32_bf16(false, Bf[cur][4], false, Bf[cur][0], (short)0, c0l, false, false);
      c1l = __builtin_amdgcn_wmma_f32_16x16x32_bf16(false, Bf[cur][4], false, Bf[cur][2], (short)0, c1l, false, false);
    }
#undef LOAD_FRAGS

    // ---- fold into running per-row argmax (C layout: row = j + 8*half, col = lane&15)
    {
      const int   code0 = kb + l16;
      const int   code1 = kb + 16 + l16;
      const float cn0   = cnorm[code0];
      const float cn1   = cnorm[code1];
#pragma unroll
      for (int j = 0; j < 8; ++j) {
        float v0 = (c0h[j] + c0l[j]) - cn0;
        if (v0 > best[j]) { best[j] = v0; bidx[j] = code0; }
        float v1 = (c1h[j] + c1l[j]) - cn1;
        if (v1 > best[j]) { best[j] = v1; bidx[j] = code1; }
      }
    }
  }

  // ---- argmax reduction across the 16 lanes sharing each row (ties -> lowest index)
#pragma unroll
  for (int s = 1; s < 16; s <<= 1) {
#pragma unroll
    for (int j = 0; j < 8; ++j) {
      float ov = __shfl_xor(best[j], s, 32);
      int   oi = __shfl_xor(bidx[j], s, 32);
      if (ov > best[j] || (ov == best[j] && oi < bidx[j])) { best[j] = ov; bidx[j] = oi; }
    }
  }
  if (l16 == 0) {
#pragma unroll
    for (int j = 0; j < 8; ++j)
      out_idx[rbase + j + 8 * half] = bidx[j];
  }

  // ---- residual update: r_m -= cb[idx_m]; 32 lanes cover D=256 (8 f32 each)
#pragma unroll
  for (int m = 0; m < 16; ++m) {
    int idx_m = __shfl(bidx[m & 7], (m >> 3) * 16, 32);
    const float* cr = cb + (size_t)idx_m * D_;
    float*       rr = res + (size_t)(rbase + m) * D_;
    float4 c0 = *(const float4*)(cr + lane * 8 + 0);
    float4 c1 = *(const float4*)(cr + lane * 8 + 4);
    float4 r0 = *(const float4*)(rr + lane * 8 + 0);
    float4 r1 = *(const float4*)(rr + lane * 8 + 4);
    r0.x -= c0.x; r0.y -= c0.y; r0.z -= c0.z; r0.w -= c0.w;
    r1.x -= c1.x; r1.y -= c1.y; r1.z -= c1.z; r1.w -= c1.w;
    *(float4*)(rr + lane * 8 + 0) = r0;
    *(float4*)(rr + lane * 8 + 4) = r1;
  }
}

// ---------------- launcher ----------------

extern "C" void kernel_launch(void* const* d_in, const int* in_sizes, int n_in,
                              void* d_out, int out_size, void* d_ws, size_t ws_size,
                              hipStream_t stream) {
  const float* x  = (const float*)d_in[0];   // [B, N, D]
  const float* cb = (const float*)d_in[1];   // [Q, K, D]

  int*   out_idx = (int*)d_out;                       // [Q, B, N] int32
  float* out_q   = (float*)d_out + (size_t)Q_ * BN;   // [B, N, D] f32 (residual during loop)
  float* cnorm   = (float*)d_ws;                      // [Q*K] 0.5*|c|^2

  const int n4 = BN * D_ / 4;  // 2,097,152 float4

  // residual <- x (lives in the quant output region)
  rvq_copy4<<<(n4 + 255) / 256, 256, 0, stream>>>((const float4*)x, (float4*)out_q, n4);

  // 0.5*|c|^2 for all Q*K codebook entries (one wave per row)
  rvq_cnorm<<<(Q_ * K_ * 32 + 255) / 256, 256, 0, stream>>>(cb, cnorm);

  // sequential RVQ stages (inherent dependency)
  for (int q = 0; q < Q_; ++q) {
    rvq_stage<<<BN / 128, 256, 0, stream>>>(
        out_q,
        cb + (size_t)q * K_ * D_,
        cnorm + (size_t)q * K_,
        out_idx + (size_t)q * BN);
  }

  // quantized = x - residual (in place)
  rvq_finalize<<<(n4 + 255) / 256, 256, 0, stream>>>((const float4*)x, (float4*)out_q, n4);
}
